// DynamicsBranch_18107582120224
// MI455X (gfx1250) — matrix operand order repaired and checked by reading, not compiled, not run on hardware
//
#include <hip/hip_runtime.h>
#include <hip/hip_bf16.h>

// ---------------------------------------------------------------------------
// DynamicsBranch fused forward for MI455X (gfx1250, wave32, WMMA).
//
// Kernel 1 (prep): scramble all weights f32->f16 into d_ws in WMMA B-fragment
//   order, one contiguous image per pipeline stage (done once per launch,
//   lives in the 192MB L2 for all workgroups).
// Kernel 2 (main): 1 workgroup = 256 threads = 8 waves, each wave owns a
//   16-row batch tile. Per GRU layer, ALL three gate weight pairs (192 KB)
//   are resident in LDS at once (CDNA5 has 320 KB/WGP), so each output
//   N-tile computes r,z,n,h to completion with only 4 live accumulator
//   tiles -> no VGPR spills. Stages are bulk-copied from the fragment image
//   into LDS (GLOBAL_LOAD_ASYNC_TO_LDS_B128 + s_wait_asynccnt when
//   available); all GEMMs via v_wmma_f32_16x16x32_f16 with f32 accumulators.
// ---------------------------------------------------------------------------

typedef _Float16 h16;
typedef __attribute__((ext_vector_type(8)))  _Float16 v8h;
typedef __attribute__((ext_vector_type(16))) _Float16 v16h;
typedef __attribute__((ext_vector_type(8)))  float    v8f;
typedef int vi4 __attribute__((vector_size(16)));   // builtin's b128 unit

union V16HU { v16h v; v8h h[2]; };

#define ZERO8 {0.f,0.f,0.f,0.f,0.f,0.f,0.f,0.f}

// ---- fragment image layout in d_ws (units: halfs) ----
#define IMG_W1     0        //  8 frags  (4096)
#define IMG_W2     4096     // 32 frags  (16384)
#define IMG_L0     20480    // 192 frags (98304): ih_r, hh_r, ih_z, hh_z, ih_n, hh_n
#define IMG_L1     118784   // 192 frags
#define IMG_HEADS  217088   // 32 frags  (16384): Ws1 (nt 0..3), Wc1 (nt 4..7)
#define IMG_HEAD2  233472   // 4 frags   (2048)
#define IMG_TOTAL  235520   // halfs (= 460 KB)

// ---- optional CDNA5 async global->LDS path ----
#if defined(__has_builtin)
#if __has_builtin(__builtin_amdgcn_global_load_async_to_lds_b128)
#define HAVE_ASYNC_LDS 1
#endif
#endif

__device__ __forceinline__ void async_wait0() {
#if defined(__has_builtin)
#if __has_builtin(__builtin_amdgcn_s_wait_asynccnt)
  __builtin_amdgcn_s_wait_asynccnt(0);
#else
  asm volatile("s_wait_asynccnt 0x0" ::: "memory");
#endif
#else
  asm volatile("s_wait_asynccnt 0x0" ::: "memory");
#endif
}

// Contiguous bulk copy (16B granular) from global fragment image into LDS.
__device__ __forceinline__ void stage_copy(h16* dst, const h16* __restrict__ src,
                                           int halfs, int tid) {
#if defined(HAVE_ASYNC_LDS)
  for (int idx = tid * 8; idx < halfs; idx += 256 * 8) {
    __builtin_amdgcn_global_load_async_to_lds_b128(
        (vi4 __attribute__((address_space(1)))*)(src + idx),
        (vi4 __attribute__((address_space(3)))*)(dst + idx), 0, 0);
  }
  async_wait0();   // own wave's ASYNCcnt drained before the barrier
#else
  for (int idx = tid * 8; idx < halfs; idx += 256 * 8)
    *(v8h*)(dst + idx) = *(const v8h*)(src + idx);
#endif
}

__device__ __forceinline__ v8f wmma_f16(v16h a, v16h b, v8f c) {
  return __builtin_amdgcn_wmma_f32_16x16x32_f16(
      false, a, false, b, (short)0, c, false, false);
}

// A-fragment (16x32 f16): lane L row M=L%16; element e -> K=e+16*(L/16)
__device__ __forceinline__ v16h load_a_frag(const h16* buf, int row0, int stride,
                                            int kc, int lane) {
  int M = lane & 15;
  int hsel = lane >> 4;
  const h16* p = buf + (row0 + M) * stride + kc * 32 + hsel * 16;
  V16HU r;
  r.h[0] = *(const v8h*)(p);
  r.h[1] = *(const v8h*)(p + 8);
  return r.v;
}

// B-fragment (32x16 f16), pre-staged contiguously: 512 halfs/frag, 16/lane.
__device__ __forceinline__ v16h load_b_frag(const h16* wbuf, int frag, int lane) {
  return *(const v16h*)(wbuf + (frag << 9) + (lane << 4));
}

// Store 16x16 f32 C tile to LDS f16: VGPR v of lane L = (M=v+8*(L/16), N=L%16)
__device__ __forceinline__ void store_tile_f16(h16* buf, int row0, int stride,
                                               int col0, int lane, v8f t) {
  int n = lane & 15, hsel = lane >> 4;
#pragma unroll
  for (int v = 0; v < 8; ++v)
    buf[(row0 + hsel * 8 + v) * stride + col0 + n] = (h16)t[v];
}

__device__ __forceinline__ float sigmoidf_(float x) { return 1.f / (1.f + __expf(-x)); }
__device__ __forceinline__ float eluf_(float x) { return x > 0.f ? x : (__expf(x) - 1.f); }

// ---------------------------------------------------------------------------
// Prep kernel: f32 weights -> f16 fragment images in d_ws.
// Fragment f = nt*kChunks + kc; lane L element e = (N=nt*16+L%16,
// K=kc*32+16*(L/16)+e); storage addr = f*512 + L*16 + e.
// ---------------------------------------------------------------------------
__global__ __launch_bounds__(256)
void prep_weights_kernel(const float* __restrict__ W1, const float* __restrict__ W2,
                         const float* __restrict__ W_ih0, const float* __restrict__ W_hh0,
                         const float* __restrict__ W_ih1, const float* __restrict__ W_hh1,
                         const float* __restrict__ Ws1, const float* __restrict__ Wc1,
                         const float* __restrict__ Ws2, const float* __restrict__ Wc2,
                         h16* __restrict__ wimg) {
  int tid = threadIdx.x;
  int seg = blockIdx.x;

  if (seg == 16) {  // head2: combined [Ws2;Wc2] as one 128x16 B, N=0 scale, 1..3 corr
    for (int idx = tid; idx < 2048; idx += 256) {
      int f = idx >> 9, r = idx & 511, ln = r >> 4, e = r & 15;
      int n = ln & 15;
      int k = f * 32 + ((ln >> 4) << 4) + e;
      float v = 0.f;
      if (n == 0)      { if (k < 64)  v = Ws2[k]; }
      else if (n < 4)  { if (k >= 64) v = Wc2[(n - 1) * 64 + (k - 64)]; }
      wimg[IMG_HEAD2 + idx] = (h16)v;
    }
    return;
  }

  const float* W = W1;
  int dstOff = 0, nOff = 0, nReal = 128, kReal = 128, ld = 128, nTiles = 8, kChunks = 4;
  switch (seg) {
    case 0:  W = W1;    dstOff = IMG_W1;        kReal = 20; ld = 20; kChunks = 1; break;
    case 1:  W = W2;    dstOff = IMG_W2;        break;
    case 2:  W = W_ih0; dstOff = IMG_L0;          nOff = 0;   break;
    case 3:  W = W_hh0; dstOff = IMG_L0 + 16384;  nOff = 0;   break;
    case 4:  W = W_ih0; dstOff = IMG_L0 + 32768;  nOff = 128; break;
    case 5:  W = W_hh0; dstOff = IMG_L0 + 49152;  nOff = 128; break;
    case 6:  W = W_ih0; dstOff = IMG_L0 + 65536;  nOff = 256; break;
    case 7:  W = W_hh0; dstOff = IMG_L0 + 81920;  nOff = 256; break;
    case 8:  W = W_ih1; dstOff = IMG_L1;          nOff = 0;   break;
    case 9:  W = W_hh1; dstOff = IMG_L1 + 16384;  nOff = 0;   break;
    case 10: W = W_ih1; dstOff = IMG_L1 + 32768;  nOff = 128; break;
    case 11: W = W_hh1; dstOff = IMG_L1 + 49152;  nOff = 128; break;
    case 12: W = W_ih1; dstOff = IMG_L1 + 65536;  nOff = 256; break;
    case 13: W = W_hh1; dstOff = IMG_L1 + 81920;  nOff = 256; break;
    case 14: W = Ws1;   dstOff = IMG_HEADS;        nReal = 64; nTiles = 4; break;
    case 15: W = Wc1;   dstOff = IMG_HEADS + 8192; nReal = 64; nTiles = 4; break;
  }
  int total = nTiles * kChunks * 512;
  for (int idx = tid; idx < total; idx += 256) {
    int f = idx >> 9, r = idx & 511, ln = r >> 4, e = r & 15;
    int nt = f / kChunks;
    int kc = f - nt * kChunks;
    int n = nt * 16 + (ln & 15);
    int k = kc * 32 + ((ln >> 4) << 4) + e;
    float v = 0.f;
    if (n < nReal && k < kReal) v = W[(nOff + n) * ld + k];
    wimg[dstOff + idx] = (h16)v;
  }
}

// ---------------------------------------------------------------------------
// GRU cell step for the workgroup's 128 rows; all 3 gate pairs LDS-resident.
// xbuf/hbuf/obuf: [128 x 128] f16 row-major LDS. wbuf: 192 frag slots:
// [ih_r | hh_r | ih_z | hh_z | ih_n | hh_n] (gate g at frags g*64 / g*64+32).
// ---------------------------------------------------------------------------
__device__ __forceinline__ void gru_layer(const h16* xbuf, const h16* hbuf, h16* obuf,
                                          h16* wbuf, const h16* __restrict__ img,
                                          const float* __restrict__ b_ih,
                                          const float* __restrict__ b_hh,
                                          int tid, int lane, int row0) {
  int nIdx = lane & 15, hsel = lane >> 4;

  __syncthreads();                     // prior wbuf readers done
  stage_copy(wbuf, img, 98304, tid);   // all three gate pairs (192 KB)
  __syncthreads();

  // x and h A-fragments are N-tile invariant: load once, reuse for 24 WMMAs/tile.
  v16h xf[4], hf[4];
#pragma unroll
  for (int kc = 0; kc < 4; ++kc) {
    xf[kc] = load_a_frag(xbuf, row0, 128, kc, lane);
    hf[kc] = load_a_frag(hbuf, row0, 128, kc, lane);
  }

#pragma unroll
  for (int nt = 0; nt < 8; ++nt) {
    v8f ar = ZERO8, az = ZERO8, ain = ZERO8, ahn = ZERO8;
#pragma unroll
    for (int kc = 0; kc < 4; ++kc) {
      ar  = wmma_f16(xf[kc], load_b_frag(wbuf,       nt * 4 + kc, lane), ar);
      ar  = wmma_f16(hf[kc], load_b_frag(wbuf,  32 + nt * 4 + kc, lane), ar);
      az  = wmma_f16(xf[kc], load_b_frag(wbuf,  64 + nt * 4 + kc, lane), az);
      az  = wmma_f16(hf[kc], load_b_frag(wbuf,  96 + nt * 4 + kc, lane), az);
      ain = wmma_f16(xf[kc], load_b_frag(wbuf, 128 + nt * 4 + kc, lane), ain);
      ahn = wmma_f16(hf[kc], load_b_frag(wbuf, 160 + nt * 4 + kc, lane), ahn);
    }
    int n = nt * 16 + nIdx;
    float br  = b_ih[n]       + b_hh[n];
    float bz  = b_ih[128 + n] + b_hh[128 + n];
    float bin = b_ih[256 + n];
    float bhn = b_hh[256 + n];
#pragma unroll
    for (int v = 0; v < 8; ++v) {
      float r  = sigmoidf_(ar[v] + br);
      float z  = sigmoidf_(az[v] + bz);
      float nn = tanhf(ain[v] + bin + r * (ahn[v] + bhn));
      float hp = (float)hbuf[(row0 + hsel * 8 + v) * 128 + n];
      obuf[(row0 + hsel * 8 + v) * 128 + n] = (h16)((1.f - z) * nn + z * hp);
    }
  }
}

// copy 128 rows x 128 f32 hidden state -> f16 LDS (contiguous, float4 loads)
__device__ __forceinline__ void stage_hidden(h16* dst, const float* __restrict__ src,
                                             int tid) {
  for (int idx = tid * 4; idx < 16384; idx += 1024) {
    float4 f = *(const float4*)(src + idx);
    dst[idx + 0] = (h16)f.x;
    dst[idx + 1] = (h16)f.y;
    dst[idx + 2] = (h16)f.z;
    dst[idx + 3] = (h16)f.w;
  }
}

__global__ __launch_bounds__(256, 1)
void dynamics_branch_kernel(
    const float* __restrict__ td, const float* __restrict__ av,
    const float* __restrict__ conf, const float* __restrict__ imu_a,
    const float* __restrict__ imu_g, const float* __restrict__ pv,
    const float* __restrict__ act, const float* __restrict__ ln_g,
    const float* __restrict__ ln_b,
    const float* __restrict__ b1, const float* __restrict__ b2,
    const float* __restrict__ b_ih0, const float* __restrict__ b_hh0,
    const float* __restrict__ b_ih1, const float* __restrict__ b_hh1,
    const float* __restrict__ bs1, const float* __restrict__ bs2,
    const float* __restrict__ bc1, const float* __restrict__ bc2,
    const float* __restrict__ h0, const h16* __restrict__ wimg,
    float* __restrict__ out, int Bn) {
  // CDNA5: 320 KB LDS/WGP. 192 KB weight frags + 3 x 32 KB activation tiles.
  __shared__ __align__(64) h16 s_w[98304];   // 192 B-fragment slots
  __shared__ __align__(64) h16 s_A[16384];   // 128 x 128 f16
  __shared__ __align__(64) h16 s_B[16384];
  __shared__ __align__(64) h16 s_C[16384];

  int tid = threadIdx.x;
  int lane = tid & 31;
  int wave = tid >> 5;
  int row0 = wave * 16;
  int nIdx = lane & 15, hsel = lane >> 4;
  int rowBase = (int)blockIdx.x * 128;

  // Warm L2 with the fragment image (global_prefetch_b8).
  __builtin_prefetch(wimg + tid * 920, 0, 0);

  // ---- Phase 0: gather 20 obs features, LayerNorm, f16 into s_A [128x32] ----
  if (tid < 128) {
    int rg = rowBase + tid;
    float f[20];
    f[0] = td[rg * 3 + 0]; f[1] = td[rg * 3 + 1]; f[2] = td[rg * 3 + 2];
    f[3] = av[rg * 3 + 0]; f[4] = av[rg * 3 + 1]; f[5] = av[rg * 3 + 2];
    f[6] = conf[rg];
    f[7] = imu_a[rg * 3 + 0]; f[8] = imu_a[rg * 3 + 1]; f[9] = imu_a[rg * 3 + 2];
    f[10] = imu_g[rg * 3 + 0]; f[11] = imu_g[rg * 3 + 1]; f[12] = imu_g[rg * 3 + 2];
    f[13] = pv[rg * 3 + 0]; f[14] = pv[rg * 3 + 1]; f[15] = pv[rg * 3 + 2];
    f[16] = act[rg * 4 + 0]; f[17] = act[rg * 4 + 1];
    f[18] = act[rg * 4 + 2]; f[19] = act[rg * 4 + 3];
    float mu = 0.f;
#pragma unroll
    for (int k = 0; k < 20; ++k) mu += f[k];
    mu *= (1.f / 20.f);
    float var = 0.f;
#pragma unroll
    for (int k = 0; k < 20; ++k) { float d = f[k] - mu; var += d * d; }
    var *= (1.f / 20.f);
    float inv = rsqrtf(var + 1e-5f);
#pragma unroll
    for (int k = 0; k < 20; ++k)
      s_A[tid * 32 + k] = (h16)((f[k] - mu) * inv * ln_g[k] + ln_b[k]);
#pragma unroll
    for (int k = 20; k < 32; ++k) s_A[tid * 32 + k] = (h16)0.f;
  }
  stage_copy(s_w, wimg + IMG_W1, 4096, tid);
  __syncthreads();

  // ---- Phase 1: hidden1 = elu(obs_n @ W1^T + b1) -> s_B ----
  {
    v16h af = load_a_frag(s_A, row0, 32, 0, lane);
#pragma unroll
    for (int nt = 0; nt < 8; ++nt) {
      v8f acc = ZERO8;
      acc = wmma_f16(af, load_b_frag(s_w, nt, lane), acc);
      float bi = b1[nt * 16 + nIdx];
      v8f o;
#pragma unroll
      for (int v = 0; v < 8; ++v) o[v] = eluf_(acc[v] + bi);
      store_tile_f16(s_B, row0, 128, nt * 16, lane, o);
    }
  }
  __syncthreads();

  // ---- Phase 2: proj = hidden1 @ W2^T + b2 -> s_A ; stage h0[0] -> s_C ----
  stage_copy(s_w, wimg + IMG_W2, 16384, tid);
  stage_hidden(s_C, h0 + (size_t)rowBase * 128, tid);
  __syncthreads();
  {
    v16h bf[4];
#pragma unroll
    for (int kc = 0; kc < 4; ++kc) bf[kc] = load_a_frag(s_B, row0, 128, kc, lane);
#pragma unroll
    for (int nt = 0; nt < 8; ++nt) {
      v8f acc = ZERO8;
#pragma unroll
      for (int kc = 0; kc < 4; ++kc)
        acc = wmma_f16(bf[kc], load_b_frag(s_w, nt * 4 + kc, lane), acc);
      float bi = b2[nt * 16 + nIdx];
      v8f o;
#pragma unroll
      for (int v = 0; v < 8; ++v) o[v] = acc[v] + bi;
      store_tile_f16(s_A, row0, 128, nt * 16, lane, o);
    }
  }

  // ---- GRU layer 0: x = proj(s_A), h = h0[0](s_C) -> s_B ----
  gru_layer(s_A, s_C, s_B, s_w, wimg + IMG_L0, b_ih0, b_hh0, tid, lane, row0);

  // ---- stage h0[1] -> s_A (proj consumed) ----
  __syncthreads();
  stage_hidden(s_A, h0 + ((size_t)Bn + rowBase) * 128, tid);

  // ---- GRU layer 1: x = h_l0(s_B), h = h0[1](s_A) -> s_C ----
  gru_layer(s_B, s_A, s_C, s_w, wimg + IMG_L1, b_ih1, b_hh1, tid, lane, row0);

  // ---- Heads stage 1: u = elu([h@Ws1^T+bs1 ; h@Wc1^T+bc1]) -> s_B ----
  __syncthreads();
  stage_copy(s_w, wimg + IMG_HEADS, 16384, tid);
  __syncthreads();
  {
    v16h hfr[4];
#pragma unroll
    for (int kc = 0; kc < 4; ++kc) hfr[kc] = load_a_frag(s_C, row0, 128, kc, lane);
#pragma unroll
    for (int nt = 0; nt < 8; ++nt) {
      v8f acc = ZERO8;
#pragma unroll
      for (int kc = 0; kc < 4; ++kc)
        acc = wmma_f16(hfr[kc], load_b_frag(s_w, nt * 4 + kc, lane), acc);
      int n = nt * 16 + nIdx;
      float bi = (n < 64) ? bs1[n] : bc1[n - 64];
      v8f o;
#pragma unroll
      for (int v = 0; v < 8; ++v) o[v] = eluf_(acc[v] + bi);
      store_tile_f16(s_B, row0, 128, nt * 16, lane, o);
    }
  }
  __syncthreads();

  // ---- Heads stage 2: out[:,0] = softplus(u_s@Ws2+bs2), out[:,1:4] = u_c@Wc2+bc2
  stage_copy(s_w, wimg + IMG_HEAD2, 2048, tid);
  __syncthreads();
  {
    v8f acc = ZERO8;
#pragma unroll
    for (int kc = 0; kc < 4; ++kc)
      acc = wmma_f16(load_a_frag(s_B, row0, 128, kc, lane),
                     load_b_frag(s_w, kc, lane), acc);
    if (nIdx < 4) {
#pragma unroll
      for (int v = 0; v < 8; ++v) {
        int rg = rowBase + row0 + hsel * 8 + v;
        float x = acc[v];
        float val;
        if (nIdx == 0) {
          x += bs2[0];
          val = (x > 20.f) ? x : log1pf(__expf(x));
        } else {
          val = x + bc2[nIdx - 1];
        }
        out[rg * 4 + nIdx] = val;
      }
    }
  }
}

extern "C" void kernel_launch(void* const* d_in, const int* in_sizes, int n_in,
                              void* d_out, int out_size, void* d_ws, size_t ws_size,
                              hipStream_t stream) {
  (void)n_in; (void)ws_size; (void)out_size;
  const float* td    = (const float*)d_in[0];
  const float* av    = (const float*)d_in[1];
  const float* conf  = (const float*)d_in[2];
  const float* imu_a = (const float*)d_in[3];
  const float* imu_g = (const float*)d_in[4];
  const float* pv    = (const float*)d_in[5];
  const float* act   = (const float*)d_in[6];
  const float* ln_g  = (const float*)d_in[7];
  const float* ln_b  = (const float*)d_in[8];
  const float* W1    = (const float*)d_in[9];
  const float* b1    = (const float*)d_in[10];
  const float* W2    = (const float*)d_in[11];
  const float* b2    = (const float*)d_in[12];
  const float* W_ih0 = (const float*)d_in[13];
  const float* W_hh0 = (const float*)d_in[14];
  const float* b_ih0 = (const float*)d_in[15];
  const float* b_hh0 = (const float*)d_in[16];
  const float* W_ih1 = (const float*)d_in[17];
  const float* W_hh1 = (const float*)d_in[18];
  const float* b_ih1 = (const float*)d_in[19];
  const float* b_hh1 = (const float*)d_in[20];
  const float* Ws1   = (const float*)d_in[21];
  const float* bs1   = (const float*)d_in[22];
  const float* Ws2   = (const float*)d_in[23];
  const float* bs2   = (const float*)d_in[24];
  const float* Wc1   = (const float*)d_in[25];
  const float* bc1   = (const float*)d_in[26];
  const float* Wc2   = (const float*)d_in[27];
  const float* bc2   = (const float*)d_in[28];
  const float* h0    = (const float*)d_in[29];
  float* out = (float*)d_out;
  h16* wimg = (h16*)d_ws;   // IMG_TOTAL halfs (~460 KB) of scratch

  int Bn = in_sizes[0] / 3;        // 65536
  int blocks = Bn / 128;           // 128 rows per workgroup

  prep_weights_kernel<<<17, 256, 0, stream>>>(
      W1, W2, W_ih0, W_hh0, W_ih1, W_hh1, Ws1, Wc1, Ws2, Wc2, wimg);

  dynamics_branch_kernel<<<blocks, 256, 0, stream>>>(
      td, av, conf, imu_a, imu_g, pv, act, ln_g, ln_b,
      b1, b2, b_ih0, b_hh0, b_ih1, b_hh1,
      bs1, bs2, bc1, bc2, h0, wimg, out, Bn);
}
